// MPNNconv_24799141167273
// MI455X (gfx1250) — matrix operand order repaired and checked by reading, not compiled, not run on hardware
//
#include <hip/hip_runtime.h>

// MPNN (NNConv + GRU) for MI455X / gfx1250, wave32.
// Roofline: dominant cost = streaming W_e each of 6 steps. Stored fp16:
// 6 * 134MB / 23.3 TB/s ~= 34us floor. W_e produced once by a
// V_WMMA_F32_16X16X32_F16 GEMM (M=16384,K=32,N=4096; K==one WMMA slice).

#define NN     8192    // nodes
#define EE     16384   // edges
#define DNI    32      // node_in
#define DEI    16      // edge_in
#define DD     64      // node_out
#define DEH    32      // edge_hidden
#define NSTEPS 6

typedef __attribute__((ext_vector_type(16))) _Float16 v16h;
typedef __attribute__((ext_vector_type(8)))  float    v8f;

union V16H { unsigned u[8]; v16h v; };
union HP   { unsigned u; _Float16 h[2]; };

// ---------------- h0 = relu(node_feats @ proj_W + proj_b) ----------------
__global__ void k_proj(const float* __restrict__ nf, const float* __restrict__ W,
                       const float* __restrict__ b, float* __restrict__ h) {
  int tid  = blockIdx.x * blockDim.x + threadIdx.x;   // NN*DD threads
  int node = tid >> 6, o = tid & 63;
  float acc = b[o];
#pragma unroll
  for (int i = 0; i < DNI; ++i)
    acc = fmaf(nf[node * DNI + i], W[i * DD + o], acc);
  h[tid] = fmaxf(acc, 0.0f);
}

// ---------------- t = relu(edge_feats @ e1_W + e1_b)  (fp16) ----------------
__global__ void k_edge1(const float* __restrict__ ef, const float* __restrict__ W,
                        const float* __restrict__ b, _Float16* __restrict__ t) {
  int tid = blockIdx.x * blockDim.x + threadIdx.x;    // EE*DEH threads
  int e = tid >> 5, k = tid & 31;
  float acc = b[k];
#pragma unroll
  for (int j = 0; j < DEI; ++j)
    acc = fmaf(ef[e * DEI + j], W[j * DEH + k], acc);
  t[tid] = (_Float16)fmaxf(acc, 0.0f);
}

// -------- stage e2_W (32x4096 f32) into wave32 WMMA B-operand layout --------
// Per n-tile (16 cols): lane half h holds col n=lane%16, K in {8h..8h+7} U
// {16+8h..16+8h+7} (VGPR v = K pair). Stored [tile][v][lane] as packed u32.
__global__ void k_stageB(const float* __restrict__ e2W, unsigned* __restrict__ staged) {
  int tid  = blockIdx.x * blockDim.x + threadIdx.x;   // 256*8*32 threads
  int lane = tid & 31, v = (tid >> 5) & 7, tile = tid >> 8;
  int n  = tile * 16 + (lane & 15);
  int k0 = (lane >> 4) * 8 + 2 * v + (v & 4) * 2;
  HP p;
  p.h[0] = (_Float16)e2W[(size_t)k0 * 4096 + n];
  p.h[1] = (_Float16)e2W[(size_t)(k0 + 1) * 4096 + n];
  staged[tid] = p.u;
}

// ---------------- convert GRU weights to fp16 (fits WGP$) ----------------
__global__ void k_cvtw(const float* __restrict__ a, const float* __restrict__ b,
                       _Float16* __restrict__ o) {
  int tid = blockIdx.x * blockDim.x + threadIdx.x;    // 2*64*192 threads
  o[tid] = (tid < DD * 3 * DD) ? (_Float16)a[tid] : (_Float16)b[tid - DD * 3 * DD];
}

// -------- W_e = t @ e2_W + e2_b  via V_WMMA_F32_16X16X32_F16, fp16 out --------
// One wave per 16-edge M-tile; A (16x32, K exactly one WMMA slice) loaded once,
// loop over 256 N-tiles. A layout per ISA: lane half h holds K in
// {8h..8h+7} U {16+8h..16+8h+7} as contiguous fp16 pairs -> dword loads.
__global__ void __launch_bounds__(256) k_edge2_wmma(
    const _Float16* __restrict__ t, const unsigned* __restrict__ stagedB,
    const float* __restrict__ b2, _Float16* __restrict__ We) {
  int wave = (blockIdx.x * blockDim.x + threadIdx.x) >> 5;  // 0..1023 = m-tile
  int lane = threadIdx.x & 31;
  int m0   = wave * 16;
  int half = lane >> 4;
  const unsigned* trow = (const unsigned*)t + (size_t)(m0 + (lane & 15)) * 16;

  V16H A;
#pragma unroll
  for (int v = 0; v < 4; ++v) A.u[v] = trow[4 * half + v];        // K 8h..8h+7
#pragma unroll
  for (int v = 4; v < 8; ++v) A.u[v] = trow[8 + 4 * half + (v - 4)]; // K 16+8h..

  for (int nt = 0; nt < 256; ++nt) {
    V16H B;
    const unsigned* bp = stagedB + nt * 256 + lane;
#pragma unroll
    for (int v = 0; v < 8; ++v) B.u[v] = bp[v * 32];
    v8f c = {};
    c = __builtin_amdgcn_wmma_f32_16x16x32_f16(false, A.v, false, B.v,
                                               (short)0, c, false, false);
    int   n    = nt * 16 + (lane & 15);
    float bias = b2[n];
#pragma unroll
    for (int r = 0; r < 8; ++r) {           // D layout: m = r + 8*half, n = lane%16
      int m = m0 + r + 8 * half;
      We[(size_t)m * 4096 + n] = (_Float16)(c[r] + bias);
    }
  }
}

// -------- NNConv message pass: wave per edge, fp16 W stream, f32 atomic scatter --------
__global__ void __launch_bounds__(256) k_msg(
    const float* __restrict__ h, const _Float16* __restrict__ We,
    const int* __restrict__ src, const int* __restrict__ dst,
    float* __restrict__ agg) {
  int e    = blockIdx.x * 8 + (threadIdx.x >> 5);
  int lane = threadIdx.x & 31;
  int s = src[e], d = dst[e];
  const unsigned* Wrow = (const unsigned*)(We + (size_t)e * 4096);
  const float*    hrow = h + (size_t)s * DD;
  __builtin_prefetch(Wrow + 2048, 0, 0);    // next edge's 8KB matrix
  float a0 = 0.0f, a1 = 0.0f;
#pragma unroll 4
  for (int i = 0; i < DD; ++i) {            // lane owns outputs o=2*lane, 2*lane+1
    float hi = hrow[i];
    HP w; w.u = Wrow[i * 32 + lane];        // 128B coalesced per wave per row
    a0 = fmaf(hi, (float)w.h[0], a0);
    a1 = fmaf(hi, (float)w.h[1], a1);
  }
  atomicAdd(&agg[(size_t)d * DD + 2 * lane], a0);
  atomicAdd(&agg[(size_t)d * DD + 2 * lane + 1], a1);
}

// -------- fused x=relu(agg+conv_b) + GRU cell; wave per node, 2 dims/lane --------
__global__ void __launch_bounds__(256) k_gru(
    const float* __restrict__ agg, const float* __restrict__ convb,
    float* __restrict__ h, const _Float16* __restrict__ Wih,
    const _Float16* __restrict__ Whh, const float* __restrict__ bih,
    const float* __restrict__ bhh) {
  __shared__ float sxh[8][128];
  int w = threadIdx.x >> 5, lane = threadIdx.x & 31;
  int n = blockIdx.x * 8 + w;
  float* sx = sxh[w];
  float* sh = sxh[w] + 64;
  sx[lane]      = fmaxf(agg[(size_t)n * DD + lane]      + convb[lane],      0.0f);
  sx[lane + 32] = fmaxf(agg[(size_t)n * DD + lane + 32] + convb[lane + 32], 0.0f);
  sh[lane]      = h[(size_t)n * DD + lane];
  sh[lane + 32] = h[(size_t)n * DD + lane + 32];
  __syncthreads();

  int d0 = 2 * lane;                               // lane owns hidden dims d0, d0+1
  float giR0 = bih[d0],       giR1 = bih[d0 + 1];
  float giZ0 = bih[64 + d0],  giZ1 = bih[64 + d0 + 1];
  float giN0 = bih[128 + d0], giN1 = bih[128 + d0 + 1];
  float ghR0 = bhh[d0],       ghR1 = bhh[d0 + 1];
  float ghZ0 = bhh[64 + d0],  ghZ1 = bhh[64 + d0 + 1];
  float ghN0 = bhh[128 + d0], ghN1 = bhh[128 + d0 + 1];
  const unsigned* Wi = (const unsigned*)Wih;       // fp16 pairs (o=2p, 2p+1)
  const unsigned* Wh = (const unsigned*)Whh;

  for (int i = 0; i < DD; ++i) {
    float xi = sx[i], hi = sh[i];
    int base = i * 96 + lane;                      // (i*192 + d0)/2
    HP r1, z1, n1, r2, z2, n2;
    r1.u = Wi[base]; z1.u = Wi[base + 32]; n1.u = Wi[base + 64];
    r2.u = Wh[base]; z2.u = Wh[base + 32]; n2.u = Wh[base + 64];
    giR0 = fmaf(xi, (float)r1.h[0], giR0); giR1 = fmaf(xi, (float)r1.h[1], giR1);
    giZ0 = fmaf(xi, (float)z1.h[0], giZ0); giZ1 = fmaf(xi, (float)z1.h[1], giZ1);
    giN0 = fmaf(xi, (float)n1.h[0], giN0); giN1 = fmaf(xi, (float)n1.h[1], giN1);
    ghR0 = fmaf(hi, (float)r2.h[0], ghR0); ghR1 = fmaf(hi, (float)r2.h[1], ghR1);
    ghZ0 = fmaf(hi, (float)z2.h[0], ghZ0); ghZ1 = fmaf(hi, (float)z2.h[1], ghZ1);
    ghN0 = fmaf(hi, (float)n2.h[0], ghN0); ghN1 = fmaf(hi, (float)n2.h[1], ghN1);
  }

  float r0 = 1.0f / (1.0f + __expf(-(giR0 + ghR0)));
  float z0 = 1.0f / (1.0f + __expf(-(giZ0 + ghZ0)));
  float t0 = tanhf(giN0 + r0 * ghN0);
  float o0 = (1.0f - z0) * t0 + z0 * sh[d0];
  float r1g = 1.0f / (1.0f + __expf(-(giR1 + ghR1)));
  float z1g = 1.0f / (1.0f + __expf(-(giZ1 + ghZ1)));
  float t1 = tanhf(giN1 + r1g * ghN1);
  float o1 = (1.0f - z1g) * t1 + z1g * sh[d0 + 1];
  h[(size_t)n * DD + d0]     = o0;
  h[(size_t)n * DD + d0 + 1] = o1;
}

extern "C" void kernel_launch(void* const* d_in, const int* in_sizes, int n_in,
                              void* d_out, int out_size, void* d_ws, size_t ws_size,
                              hipStream_t stream) {
  (void)in_sizes; (void)n_in; (void)out_size; (void)ws_size;
  const float* node_feats = (const float*)d_in[0];
  const float* edge_feats = (const float*)d_in[1];
  const int*   src        = (const int*)d_in[2];
  const int*   dst        = (const int*)d_in[3];
  const float* projW      = (const float*)d_in[4];
  const float* projb      = (const float*)d_in[5];
  const float* e1W        = (const float*)d_in[6];
  const float* e1b        = (const float*)d_in[7];
  const float* e2W        = (const float*)d_in[8];
  const float* e2b        = (const float*)d_in[9];
  const float* convb      = (const float*)d_in[10];
  const float* Wih        = (const float*)d_in[11];
  const float* Whh        = (const float*)d_in[12];
  const float* bih        = (const float*)d_in[13];
  const float* bhh        = (const float*)d_in[14];
  float* h = (float*)d_out;

  // workspace carve-up (~138 MB)
  char* ws = (char*)d_ws;
  _Float16* We      = (_Float16*)ws; ws += (size_t)EE * 4096 * 2;  // 134 MB
  float*    agg     = (float*)ws;    ws += (size_t)NN * DD * 4;    // 2 MB
  _Float16* t16     = (_Float16*)ws; ws += (size_t)EE * DEH * 2;   // 1 MB
  unsigned* stagedB = (unsigned*)ws; ws += (size_t)256 * 8 * 32 * 4;
  _Float16* gruW16  = (_Float16*)ws; ws += (size_t)2 * DD * 3 * DD * 2;

  // one-time setup
  k_proj      <<<2048, 256, 0, stream>>>(node_feats, projW, projb, h);
  k_edge1     <<<2048, 256, 0, stream>>>(edge_feats, e1W, e1b, t16);
  k_stageB    <<<256,  256, 0, stream>>>(e2W, stagedB);
  k_cvtw      <<<96,   256, 0, stream>>>(Wih, Whh, gruW16);
  k_edge2_wmma<<<128,  256, 0, stream>>>(t16, stagedB, e2b, We);

  // 6 message-passing steps
  for (int s = 0; s < NSTEPS; ++s) {
    (void)hipMemsetAsync(agg, 0, (size_t)NN * DD * 4, stream);
    k_msg<<<2048, 256, 0, stream>>>(h, We, src, dst, agg);
    k_gru<<<1024, 256, 0, stream>>>(agg, convb, h, gruW16, gruW16 + DD * 3 * DD,
                                    bih, bhh);
  }
}
